// Model_40630390620510
// MI455X (gfx1250) — compile-verified, matrix operand order checked by reading, not run on hardware
//
#include <hip/hip_runtime.h>
#include <math.h>

typedef __attribute__((ext_vector_type(16))) _Float16 v16h;
typedef __attribute__((ext_vector_type(8)))  float    v8f;

#define TK 32

// ---------------------------------------------------------------------------
// f32 -> f16 operand staging (halves repeated global traffic for GEMM tiles,
// enables 16B/lane async Global->LDS loads)
// ---------------------------------------------------------------------------
__global__ void tof16_k(const float* __restrict__ in, _Float16* __restrict__ out, long n)
{
  long i = (long)blockIdx.x * blockDim.x + threadIdx.x;
  if (i < n) out[i] = (_Float16)in[i];
}

// ---------------------------------------------------------------------------
// WMMA GEMM: C[M,N] = act(A[M,K] @ W[N,K]^T + bias) (+ resid)
// A,W are f16. block = 256 threads = 8 waves; block tile 64x64; wave tile
// 16x32 (2 WMMAs / K-step).
// Interior blocks: double-buffered software pipeline -- each K-tile (4KB per
// operand = 256 lanes x 16B) is fetched with one global_load_async_to_lds_b128
// per operand for the *next* iteration while WMMAs consume the current one;
// completion tracked with ASYNCcnt (s_wait_asynccnt before the publishing
// barrier). Edge blocks fall back to guarded scalar staging.
// ---------------------------------------------------------------------------
__global__ __launch_bounds__(256) void swin_gemm_wmma(
    const _Float16* __restrict__ A, const _Float16* __restrict__ W,
    const float* __restrict__ bias, const float* __restrict__ resid,
    float* __restrict__ C, int M, int N, int K, int act)
{
  __shared__ __align__(16) _Float16 As[2][64][TK + 8];
  __shared__ __align__(16) _Float16 Bs[2][64][TK + 8];
  const int m0 = blockIdx.y * 64, n0 = blockIdx.x * 64;
  const int tid = threadIdx.x, lane = tid & 31, wid = tid >> 5;
  const int wm = (wid >> 1) * 16;   // 0,16,32,48
  const int wn = (wid & 1) * 32;    // 0,32
  const int row = lane & 15, grp = lane >> 4;
  v8f acc0 = {};
  v8f acc1 = {};

  const bool interior = (m0 + 64 <= M) && (n0 + 64 <= N) && (K % TK == 0);
  // per-thread 16B slot within the 64x32 tile
  const int tr = tid >> 2;          // 0..63
  const int tc = (tid & 3) * 8;     // 0,8,16,24

  auto prefetch = [&](int kt, int bufi) {
    const _Float16* ga = A + (size_t)(m0 + tr) * K + kt * TK + tc;
    const _Float16* gb = W + (size_t)(n0 + tr) * K + kt * TK + tc;
    unsigned la = (unsigned)(size_t)&As[bufi][tr][tc];
    unsigned lb = (unsigned)(size_t)&Bs[bufi][tr][tc];
    asm volatile("global_load_async_to_lds_b128 %0, %1, off"
                 :: "v"(la), "v"(ga) : "memory");
    asm volatile("global_load_async_to_lds_b128 %0, %1, off"
                 :: "v"(lb), "v"(gb) : "memory");
  };

  if (interior) {
    const int nk = K / TK;
    prefetch(0, 0);
    for (int i = 0; i < nk; ++i) {
      int cur = i & 1;
      asm volatile("s_wait_asynccnt 0x0" ::: "memory");
      __syncthreads();                      // publish buf[cur] to all waves
      if (i + 1 < nk) prefetch(i + 1, 1 - cur); // overlap next copy with WMMA
      v16h a, b0, b1;
#pragma unroll
      for (int e = 0; e < 16; ++e) {
        int vv = e >> 1;
        int kk = 2 * vv + (e & 1) + 8 * grp + ((vv >= 4) ? 8 : 0);
        a[e]  = As[cur][wm + row][kk];
        b0[e] = Bs[cur][wn + row][kk];
        b1[e] = Bs[cur][wn + 16 + row][kk];
      }
      acc0 = __builtin_amdgcn_wmma_f32_16x16x32_f16(false, a, false, b0, (short)0, acc0, false, false);
      acc1 = __builtin_amdgcn_wmma_f32_16x16x32_f16(false, a, false, b1, (short)0, acc1, false, false);
      __syncthreads();                      // reads done before buf reuse
    }
  } else {
    for (int k0 = 0; k0 < K; k0 += TK) {
      for (int i = tid; i < 64 * TK; i += 256) {
        int r = i / TK, c = i % TK;
        int gm = m0 + r, gn = n0 + r, gk = k0 + c;
        As[0][r][c] = (gm < M && gk < K) ? A[(size_t)gm * K + gk] : (_Float16)0.f;
        Bs[0][r][c] = (gn < N && gk < K) ? W[(size_t)gn * K + gk] : (_Float16)0.f;
      }
      __syncthreads();
      v16h a, b0, b1;
#pragma unroll
      for (int e = 0; e < 16; ++e) {
        int vv = e >> 1;
        int kk = 2 * vv + (e & 1) + 8 * grp + ((vv >= 4) ? 8 : 0);
        a[e]  = As[0][wm + row][kk];
        b0[e] = Bs[0][wn + row][kk];
        b1[e] = Bs[0][wn + 16 + row][kk];
      }
      acc0 = __builtin_amdgcn_wmma_f32_16x16x32_f16(false, a, false, b0, (short)0, acc0, false, false);
      acc1 = __builtin_amdgcn_wmma_f32_16x16x32_f16(false, a, false, b1, (short)0, acc1, false, false);
      __syncthreads();
    }
  }

  // C/D layout: n = lane&15 ; m = r + 8*(lane>>4)
  const int cn = lane & 15, cm = 8 * grp;
#pragma unroll
  for (int r = 0; r < 8; ++r) {
    int gm = m0 + wm + cm + r;
    if (gm >= M) continue;
    int gn = n0 + wn + cn;
    if (gn < N) {
      float x = acc0[r] + (bias ? bias[gn] : 0.f);
      if (act == 1) x = 0.5f * x * (1.f + erff(x * 0.70710678f));
      if (resid) x += resid[(size_t)gm * N + gn];
      C[(size_t)gm * N + gn] = x;
    }
    int gn2 = gn + 16;
    if (gn2 < N) {
      float x = acc1[r] + (bias ? bias[gn2] : 0.f);
      if (act == 1) x = 0.5f * x * (1.f + erff(x * 0.70710678f));
      if (resid) x += resid[(size_t)gm * N + gn2];
      C[(size_t)gm * N + gn2] = x;
    }
  }
}

static _Float16* g_hA = nullptr;
static _Float16* g_hW = nullptr;

static inline void gemm(hipStream_t s, const float* A, const float* W, const float* bias,
                        const float* resid, float* C, int M, int N, int K, int act)
{
  long na = (long)M * K, nw = (long)N * K;
  tof16_k<<<(unsigned)((na + 255) / 256), 256, 0, s>>>(A, g_hA, na);
  tof16_k<<<(unsigned)((nw + 255) / 256), 256, 0, s>>>(W, g_hW, nw);
  dim3 g((N + 63) / 64, (M + 63) / 64);
  swin_gemm_wmma<<<g, 256, 0, s>>>(g_hA, g_hW, bias, resid, C, M, N, K, act);
}

// ---------------------------------------------------------------------------
// Patch embed: 4x4/4 conv, NCHW in -> [B, 3136, 96] token-major out
// ---------------------------------------------------------------------------
__global__ void patch_embed_k(const float* __restrict__ x, const float* __restrict__ w,
                              const float* __restrict__ b, float* __restrict__ out)
{
  int i = blockIdx.x * blockDim.x + threadIdx.x;
  const int L = 3136, Cc = 96;
  if (i >= 8 * L * Cc) return;
  int co = i % Cc;
  int l  = (i / Cc) % L;
  int bb = i / (Cc * L);
  int oh = l / 56, ow = l % 56;
  float s = b[co];
  for (int ci = 0; ci < 3; ++ci)
    for (int kh = 0; kh < 4; ++kh)
      for (int kw = 0; kw < 4; ++kw)
        s += x[(((size_t)bb * 3 + ci) * 224 + oh * 4 + kh) * 224 + ow * 4 + kw] *
             w[((co * 3 + ci) * 4 + kh) * 4 + kw];
  out[i] = s;
}

// ---------------------------------------------------------------------------
// Row LayerNorm (one thread per row; C <= 768)
// ---------------------------------------------------------------------------
__global__ void layernorm_k(const float* __restrict__ in, float* __restrict__ out,
                            const float* __restrict__ w, const float* __restrict__ b,
                            int rows, int C)
{
  int r = blockIdx.x * blockDim.x + threadIdx.x;
  if (r >= rows) return;
  const float* p = in + (size_t)r * C;
  float m = 0.f;
  for (int c = 0; c < C; ++c) m += p[c];
  m /= (float)C;
  float v = 0.f;
  for (int c = 0; c < C; ++c) { float d = p[c] - m; v += d * d; }
  v /= (float)C;
  float inv = rsqrtf(v + 1e-5f);
  float* o = out + (size_t)r * C;
  for (int c = 0; c < C; ++c) o[c] = (p[c] - m) * inv * w[c] + b[c];
}

// ---------------------------------------------------------------------------
// Window partition with cyclic shift: x[B,res,res,C] -> xw[B*nW,49,C]
// ---------------------------------------------------------------------------
__global__ void winpart_k(const float* __restrict__ x, float* __restrict__ xw,
                          int Bn, int res, int C, int win, int shift)
{
  long total = (long)Bn * res * res * C;
  long i = (long)blockIdx.x * blockDim.x + threadIdx.x;
  if (i >= total) return;
  int c = i % C;
  long t2 = i / C;
  int tok = t2 % (win * win);
  int bw  = t2 / (win * win);
  int nWr = res / win;
  int b = bw / (nWr * nWr);
  int w = bw % (nWr * nWr);
  int r  = (w / nWr) * win + tok / win;
  int cc = (w % nWr) * win + tok % win;
  int sr = (r + shift) % res;
  int sc = (cc + shift) % res;
  xw[i] = x[(((size_t)b * res + sr) * res + sc) * C + c];
}

// ---------------------------------------------------------------------------
// Window reverse + un-shift + residual add: out = resid + reverse(aw)
// ---------------------------------------------------------------------------
__global__ void winrev_k(const float* __restrict__ aw, const float* __restrict__ resid,
                         float* __restrict__ out, int Bn, int res, int C, int win, int shift)
{
  long total = (long)Bn * res * res * C;
  long i = (long)blockIdx.x * blockDim.x + threadIdx.x;
  if (i >= total) return;
  int c = i % C;
  long t2 = i / C;
  int pos = t2 % (res * res);
  int b   = t2 / (res * res);
  int r = pos / res, cc = pos % res;
  int ri = (r - shift + res) % res;
  int cj = (cc - shift + res) % res;
  int nWr = res / win;
  int bw  = b * nWr * nWr + (ri / win) * nWr + (cj / win);
  int tok = (ri % win) * win + (cj % win);
  out[i] = resid[i] + aw[((size_t)bw * win * win + tok) * C + c];
}

// ---------------------------------------------------------------------------
// qkv bias = cat(q_bias, 0, v_bias)
// ---------------------------------------------------------------------------
__global__ void qkvbias_k(const float* __restrict__ qb, const float* __restrict__ vb,
                          float* __restrict__ out, int C)
{
  int i = blockIdx.x * blockDim.x + threadIdx.x;
  if (i >= 3 * C) return;
  out[i] = (i < C) ? qb[i] : (i < 2 * C ? 0.f : vb[i - 2 * C]);
}

// ---------------------------------------------------------------------------
// CPB MLP: (2 -> 512 relu -> nH), then 16*sigmoid.  out[(2w-1)^2][nH]
// ---------------------------------------------------------------------------
__global__ void cpb_k(const float* __restrict__ ct, const float* __restrict__ w0,
                      const float* __restrict__ b0, const float* __restrict__ w1,
                      float* __restrict__ out, int nH)
{
  int i = blockIdx.x * blockDim.x + threadIdx.x;
  const int P = 169; // (2*7-1)^2
  if (i >= P * nH) return;
  int p = i / nH, h = i % nH;
  float x0 = ct[p * 2 + 0], x1 = ct[p * 2 + 1];
  float acc = 0.f;
  for (int j = 0; j < 512; ++j) {
    float hv = fmaxf(0.f, w0[j * 2 + 0] * x0 + w0[j * 2 + 1] * x1 + b0[j]);
    acc += w1[h * 512 + j] * hv;
  }
  out[p * nH + h] = 16.f / (1.f + expf(-acc));
}

// ---------------------------------------------------------------------------
// Cosine window attention core (N=49, head_dim=32). One block per (window,head).
// ---------------------------------------------------------------------------
__global__ __launch_bounds__(64) void winattn_k(
    const float* __restrict__ qkv, float* __restrict__ out,
    const float* __restrict__ lscale, const float* __restrict__ cpb,
    const int* __restrict__ rpi, const float* __restrict__ mask,
    int nWimg, int nH, int C)
{
  const int N = 49, D = 32;
  int wi = blockIdx.x / nH, h = blockIdx.x % nH;
  __shared__ float q[49][33], k[49][33], v[49][33], pr[49][50];
  int t = threadIdx.x;
  if (t < N) {
    size_t rb = (size_t)(wi * N + t) * (3 * C);
    float nq = 0.f, nk = 0.f;
    for (int d = 0; d < D; ++d) {
      float qq = qkv[rb + h * D + d];
      float kk = qkv[rb + C + h * D + d];
      v[t][d] = qkv[rb + 2 * C + h * D + d];
      q[t][d] = qq; k[t][d] = kk;
      nq += qq * qq; nk += kk * kk;
    }
    nq = fmaxf(sqrtf(nq), 1e-12f);
    nk = fmaxf(sqrtf(nk), 1e-12f);
    for (int d = 0; d < D; ++d) { q[t][d] /= nq; k[t][d] /= nk; }
  }
  __syncthreads();
  if (t < N) {
    float scale = expf(fminf(lscale[h], 4.6051702f)); // clamp at log(100)
    float mx = -1e30f;
    for (int j = 0; j < N; ++j) {
      float ds = 0.f;
      for (int d = 0; d < D; ++d) ds += q[t][d] * k[j][d];
      float a = ds * scale + cpb[rpi[t * N + j] * nH + h];
      if (mask) a += mask[(size_t)(wi % nWimg) * N * N + t * N + j];
      pr[t][j] = a;
      if (a > mx) mx = a;
    }
    float s = 0.f;
    for (int j = 0; j < N; ++j) { float e = expf(pr[t][j] - mx); pr[t][j] = e; s += e; }
    float inv = 1.f / s;
    size_t ob = (size_t)(wi * N + t) * C + h * D;
    for (int d = 0; d < D; ++d) {
      float o = 0.f;
      for (int j = 0; j < N; ++j) o += pr[t][j] * v[j][d];
      out[ob + d] = o * inv;
    }
  }
}

// ---------------------------------------------------------------------------
// Patch-merge gather: x[B,res,res,C] -> out[B,(res/2)^2, 4C] in quadrant order
// ---------------------------------------------------------------------------
__global__ void merge_k(const float* __restrict__ x, float* __restrict__ out,
                        int Bn, int res, int C)
{
  int half = res / 2;
  long total = (long)Bn * half * half * 4 * C;
  long i = (long)blockIdx.x * blockDim.x + threadIdx.x;
  if (i >= total) return;
  int c4  = i % (4 * C);
  long t2 = i / (4 * C);
  int pos = t2 % (half * half);
  int b   = t2 / (half * half);
  int quad = c4 / C, c = c4 % C;
  int r = pos / half, cc = pos % half;
  int sr = 2 * r + (quad & 1);
  int sc = 2 * cc + (quad >> 1);
  out[i] = x[(((size_t)b * res + sr) * res + sc) * C + c];
}

__global__ void meanpool_k(const float* __restrict__ x, float* __restrict__ out,
                           int Bn, int L, int C)
{
  int i = blockIdx.x * blockDim.x + threadIdx.x;
  if (i >= Bn * C) return;
  int b = i / C, c = i % C;
  float s = 0.f;
  for (int l = 0; l < L; ++l) s += x[((size_t)b * L + l) * C + c];
  out[i] = s / (float)L;
}

// ---------------------------------------------------------------------------
// Host orchestration
// ---------------------------------------------------------------------------
extern "C" void kernel_launch(void* const* d_in, const int* in_sizes, int n_in,
                              void* d_out, int out_size, void* d_ws, size_t ws_size,
                              hipStream_t stream)
{
  (void)in_sizes; (void)n_in; (void)out_size; (void)ws_size;
  auto F = [&](int i) { return (const float*)d_in[i]; };
  auto I = [&](int i) { return (const int*)d_in[i]; };

  const float* x_in = F(0);
  int idx = 1;
  // params leaves, sorted-key order
  const float* head_b = F(idx++);
  const float* head_w = F(idx++);
  struct BP { const float *cpb_b0,*cpb_w0,*cpb_w1,*fc1_b,*fc1_w,*fc2_b,*fc2_w,
                        *lscale,*n1_b,*n1_w,*n2_b,*n2_w,*proj_b,*proj_w,
                        *q_bias,*qkv_w,*v_bias; } bp[12];
  for (int b = 0; b < 12; ++b) {
    bp[b].cpb_b0 = F(idx++); bp[b].cpb_w0 = F(idx++); bp[b].cpb_w1 = F(idx++);
    bp[b].fc1_b  = F(idx++); bp[b].fc1_w  = F(idx++);
    bp[b].fc2_b  = F(idx++); bp[b].fc2_w  = F(idx++);
    bp[b].lscale = F(idx++);
    bp[b].n1_b = F(idx++); bp[b].n1_w = F(idx++);
    bp[b].n2_b = F(idx++); bp[b].n2_w = F(idx++);
    bp[b].proj_b = F(idx++); bp[b].proj_w = F(idx++);
    bp[b].q_bias = F(idx++); bp[b].qkv_w = F(idx++); bp[b].v_bias = F(idx++);
  }
  struct MP { const float *norm_b,*norm_w,*red_w; } mp[3];
  for (int m = 0; m < 3; ++m) { mp[m].norm_b = F(idx++); mp[m].norm_w = F(idx++); mp[m].red_w = F(idx++); }
  const float* fin_norm_b = F(idx++);
  const float* fin_norm_w = F(idx++);
  const float* pe_b      = F(idx++);
  const float* pe_norm_b = F(idx++);
  const float* pe_norm_w = F(idx++);
  const float* pe_w      = F(idx++);
  // consts leaves, sorted-key order; shift/win scalars skipped (known statically)
  struct BC { const float* coords; const float* mask; const int* rpi; } bc[12];
  const int has_mask[12] = {0,1,0,1,0,1,0,1,0,1,0,0};
  for (int b = 0; b < 12; ++b) {
    bc[b].coords = F(idx++);
    bc[b].mask = has_mask[b] ? F(idx++) : nullptr;
    bc[b].rpi = I(idx++);
    idx += 2; // shift, win
  }

  // workspace carve (floats, 16B aligned for b128 async loads)
  float* ws = (float*)d_ws;
  size_t off = 0;
  auto carve = [&](size_t n) { float* p = ws + off; off += (n + 7) & ~(size_t)7; return p; };
  float* xb   = carve(2408448);   // activations [rows, C]
  float* tmp  = carve(2408448);   // layernorm output
  float* xw   = carve(2408448);   // windowed tokens / merge gather
  float* qkvb = carve(7225344);   // qkv projections
  float* attn = carve(2408448);   // attention outputs
  float* proj = carve(2408448);   // proj / merge gemm outputs
  float* hid  = carve(9633792);   // MLP hidden
  float* cpb  = carve(8192);      // 16*sigmoid(CPB) table
  float* qb3  = carve(4096);      // qkv bias concat
  float* pool = carve(8192);      // mean-pooled features
  g_hA = (_Float16*)carve(4816896); // f16 staging: A operand (9,633,792 halves)
  g_hW = (_Float16*)carve(1179648); // f16 staging: W operand (2,359,296 halves)

  const int Bn = 8;
  // --- patch embed + LN ---
  {
    int total = Bn * 3136 * 96;
    patch_embed_k<<<(total + 255) / 256, 256, 0, stream>>>(x_in, pe_w, pe_b, xb);
    layernorm_k<<<(Bn * 3136 + 255) / 256, 256, 0, stream>>>(xb, xb, pe_norm_w, pe_norm_b, Bn * 3136, 96);
  }

  const int DEP[4] = {2, 2, 6, 2};
  const int HEA[4] = {3, 6, 12, 24};
  int res = 56, C = 96, bi = 0;
  for (int il = 0; il < 4; ++il) {
    int nH = HEA[il];
    int L = res * res, rows = Bn * L;
    int nWr = res / 7, nWimg = nWr * nWr, totWin = Bn * nWimg;
    for (int ib = 0; ib < DEP[il]; ++ib, ++bi) {
      int shift = (ib % 2 == 1 && res > 7) ? 3 : 0;
      const BP& p = bp[bi];
      const BC& c = bc[bi];
      long totalE = (long)rows * C;
      // attention branch
      layernorm_k<<<(rows + 255) / 256, 256, 0, stream>>>(xb, tmp, p.n1_w, p.n1_b, rows, C);
      winpart_k<<<(totalE + 255) / 256, 256, 0, stream>>>(tmp, xw, Bn, res, C, 7, shift);
      qkvbias_k<<<(3 * C + 255) / 256, 256, 0, stream>>>(p.q_bias, p.v_bias, qb3, C);
      gemm(stream, xw, p.qkv_w, qb3, nullptr, qkvb, rows, 3 * C, C, 0);
      cpb_k<<<(169 * nH + 63) / 64, 64, 0, stream>>>(c.coords, p.cpb_w0, p.cpb_b0, p.cpb_w1, cpb, nH);
      winattn_k<<<totWin * nH, 64, 0, stream>>>(qkvb, attn, p.lscale, cpb, c.rpi, c.mask, nWimg, nH, C);
      gemm(stream, attn, p.proj_w, p.proj_b, nullptr, proj, rows, C, C, 0);
      winrev_k<<<(totalE + 255) / 256, 256, 0, stream>>>(proj, xb, xb, Bn, res, C, 7, shift);
      // MLP branch
      layernorm_k<<<(rows + 255) / 256, 256, 0, stream>>>(xb, tmp, p.n2_w, p.n2_b, rows, C);
      gemm(stream, tmp, p.fc1_w, p.fc1_b, nullptr, hid, rows, 4 * C, C, 1 /*gelu*/);
      gemm(stream, hid, p.fc2_w, p.fc2_b, xb /*residual*/, xb, rows, C, 4 * C, 0);
    }
    if (il < 3) {
      int half = res / 2, rows2 = Bn * half * half;
      long tot = (long)rows2 * 4 * C;
      merge_k<<<(tot + 255) / 256, 256, 0, stream>>>(xb, xw, Bn, res, C);
      gemm(stream, xw, mp[il].red_w, nullptr, nullptr, proj, rows2, 2 * C, 4 * C, 0);
      layernorm_k<<<(rows2 + 255) / 256, 256, 0, stream>>>(proj, xb, mp[il].norm_w, mp[il].norm_b, rows2, 2 * C);
      res = half; C *= 2;
    }
  }

  // final norm -> mean pool -> head
  layernorm_k<<<(Bn * 49 + 255) / 256, 256, 0, stream>>>(xb, tmp, fin_norm_w, fin_norm_b, Bn * 49, 768);
  meanpool_k<<<(Bn * 768 + 255) / 256, 256, 0, stream>>>(tmp, pool, Bn, 49, 768);
  gemm(stream, pool, head_w, head_b, nullptr, (float*)d_out, Bn, 1000, 768, 0);
}
